// GlobalMlpBlock_26293789786458
// MI455X (gfx1250) — compile-verified, matrix operand order checked by reading, not run on hardware
//
#include <hip/hip_runtime.h>
#include <hip/hip_bf16.h>

typedef __attribute__((ext_vector_type(16))) __bf16 v16bf;
typedef __attribute__((ext_vector_type(8)))  float  v8f;
typedef unsigned short u16;
typedef unsigned int   u32;

union FragU { v16bf v; u32 u[8]; uint4 q[2]; };

__device__ __forceinline__ u16 f32_to_bf16(float f) {
    u32 u = __float_as_uint(f);
    u += 0x7FFFu + ((u >> 16) & 1u);   // round-to-nearest-even
    return (u16)(u >> 16);
}

__device__ __forceinline__ float gelu_exact(float v) {
    return 0.5f * v * (1.0f + erff(v * 0.70710678118654752f));
}

// ---------------------------------------------------------------------------
// LayerNorm of input rows (dim=1024), emitting bf16(x) and bf16(LN(x))
// ---------------------------------------------------------------------------
__global__ __launch_bounds__(256)
void ln_in_kernel(const float* __restrict__ x,
                  const float* __restrict__ gamma,
                  const float* __restrict__ beta,
                  u16* __restrict__ xb, u16* __restrict__ xn) {
    const int row = blockIdx.x;
    const int tid = threadIdx.x;
    const float4 xv = ((const float4*)(x + (size_t)row * 1024))[tid];

    __shared__ float ssum[256], ssq[256];
    ssum[tid] = xv.x + xv.y + xv.z + xv.w;
    ssq[tid]  = xv.x*xv.x + xv.y*xv.y + xv.z*xv.z + xv.w*xv.w;
    __syncthreads();
    for (int o = 128; o > 0; o >>= 1) {
        if (tid < o) { ssum[tid] += ssum[tid+o]; ssq[tid] += ssq[tid+o]; }
        __syncthreads();
    }
    const float mu   = ssum[0] * (1.0f/1024.0f);
    const float var  = ssq[0]  * (1.0f/1024.0f) - mu*mu;
    const float rstd = rsqrtf(var + 1e-5f);

    const float vals[4] = {xv.x, xv.y, xv.z, xv.w};
    #pragma unroll
    for (int j = 0; j < 4; ++j) {
        const int col = tid*4 + j;
        const float nv = (vals[j]-mu)*rstd*gamma[col] + beta[col];
        xb[(size_t)row*1024 + col] = f32_to_bf16(vals[j]);
        xn[(size_t)row*1024 + col] = f32_to_bf16(nv);
    }
}

// ---------------------------------------------------------------------------
// Final LayerNorm: f32 in, f32 out
// ---------------------------------------------------------------------------
__global__ __launch_bounds__(256)
void ln_out_kernel(const float* __restrict__ yp,
                   const float* __restrict__ gamma,
                   const float* __restrict__ beta,
                   float* __restrict__ out) {
    const int row = blockIdx.x;
    const int tid = threadIdx.x;
    const float4 xv = ((const float4*)(yp + (size_t)row * 1024))[tid];

    __shared__ float ssum[256], ssq[256];
    ssum[tid] = xv.x + xv.y + xv.z + xv.w;
    ssq[tid]  = xv.x*xv.x + xv.y*xv.y + xv.z*xv.z + xv.w*xv.w;
    __syncthreads();
    for (int o = 128; o > 0; o >>= 1) {
        if (tid < o) { ssum[tid] += ssum[tid+o]; ssq[tid] += ssq[tid+o]; }
        __syncthreads();
    }
    const float mu   = ssum[0] * (1.0f/1024.0f);
    const float var  = ssq[0]  * (1.0f/1024.0f) - mu*mu;
    const float rstd = rsqrtf(var + 1e-5f);

    const float vals[4] = {xv.x, xv.y, xv.z, xv.w};
    float4 ov;
    ov.x = (vals[0]-mu)*rstd*gamma[tid*4+0] + beta[tid*4+0];
    ov.y = (vals[1]-mu)*rstd*gamma[tid*4+1] + beta[tid*4+1];
    ov.z = (vals[2]-mu)*rstd*gamma[tid*4+2] + beta[tid*4+2];
    ov.w = (vals[3]-mu)*rstd*gamma[tid*4+3] + beta[tid*4+3];
    ((float4*)(out + (size_t)row * 1024))[tid] = ov;
}

// ---------------------------------------------------------------------------
// f32 [R,C] -> bf16 transposed [C,R]  (weights: done once, cost negligible)
// ---------------------------------------------------------------------------
__global__ __launch_bounds__(256)
void convert_transpose_kernel(const float* __restrict__ in, u16* __restrict__ out,
                              int R, int C) {
    __shared__ u16 t[32][33];
    const int c0 = blockIdx.x * 32;
    const int r0 = blockIdx.y * 32;
    const int tx = threadIdx.x, ty = threadIdx.y;   // (32, 8)
    #pragma unroll
    for (int i = 0; i < 4; ++i) {
        const int r = ty + i * 8;
        t[r][tx] = f32_to_bf16(in[(size_t)(r0 + r) * C + c0 + tx]);
    }
    __syncthreads();
    #pragma unroll
    for (int i = 0; i < 4; ++i) {
        const int r = ty + i * 8;
        out[(size_t)(c0 + r) * R + r0 + tx] = t[tx][r];
    }
}

// ---------------------------------------------------------------------------
// bf16 [R,C] -> bf16 transposed [C,R], batched (for y2 -> y2^T per batch)
// ---------------------------------------------------------------------------
__global__ __launch_bounds__(256)
void transpose_u16_kernel(const u16* __restrict__ in, u16* __restrict__ out,
                          int R, int C, long long sIn, long long sOut) {
    __shared__ u16 t[32][33];
    in  += (size_t)blockIdx.z * sIn;
    out += (size_t)blockIdx.z * sOut;
    const int c0 = blockIdx.x * 32;
    const int r0 = blockIdx.y * 32;
    const int tx = threadIdx.x, ty = threadIdx.y;   // (32, 8)
    #pragma unroll
    for (int i = 0; i < 4; ++i) {
        const int r = ty + i * 8;
        t[r][tx] = in[(size_t)(r0 + r) * C + c0 + tx];
    }
    __syncthreads();
    #pragma unroll
    for (int i = 0; i < 4; ++i) {
        const int r = ty + i * 8;
        out[(size_t)(c0 + r) * R + r0 + tx] = t[tx][r];
    }
}

// ---------------------------------------------------------------------------
// bf16 GEMM: C[M,N] = A[M,K] x B^T  where BOTH operands are K-major:
//   A is [M,K] row-major, B is [N,K] row-major.
// Block tile 128x128, 8 waves (4 in M x 2 in N), wave = 32x64 = 2x4 WMMA tiles.
// A/B tiles staged into double-buffered LDS with CDNA5 async global->LDS
// (global_load_async_to_lds_b128, ASYNCcnt), overlapping next-tile HBM
// latency with current-tile WMMA. Fragment loads are pure ds_load_b128.
// EPI: 0 = store bf16(acc)              (scores)
//      1 = store bf16(gelu(acc+bias))   (GEMM1)
//      2 = store bf16(acc+bias)         (GEMM2)
//      3 = store f32(acc + resid)       (GEMM3, residual add)
// ---------------------------------------------------------------------------
template<int EPI>
__global__ __launch_bounds__(256)
void gemm_bf16_kernel(const u16* __restrict__ A, int lda, long long sA,
                      const u16* __restrict__ B, int ldb, long long sB,
                      void* __restrict__ Cv, int ldc, long long sC,
                      const float* __restrict__ bias,
                      const float* __restrict__ resid, long long sR,
                      int K) {
    const int tid  = threadIdx.x;
    const int lane = tid & 31;
    const int wid  = tid >> 5;
    const int wm   = wid >> 1;          // 0..3  (M direction)
    const int wn   = wid & 1;           // 0..1  (N direction)
    const int hi   = lane >> 4;         // 0/1 (half-wave)
    const int lo   = lane & 15;

    const int mBase = blockIdx.y * 128;
    const int nBase = blockIdx.x * 128;
    const int batch = blockIdx.z;
    A += (size_t)batch * sA;
    B += (size_t)batch * sB;
    const size_t cbase = (size_t)batch * sC;

    // Double-buffered tiles: 128 rows x 32 K-elems, row pitch 40 u16 (80B).
    __shared__ __align__(16) u16 As[2][128 * 40];
    __shared__ __align__(16) u16 Bs[2][128 * 40];

    // Staging assignment: 2 threads per row, 16 u16 (32B) each = 2 async b128.
    const int rS  = tid >> 1;
    const int seg = (tid & 1) << 4;
    const u16* gA0 = A + (size_t)(mBase + rS) * lda + seg;
    const u16* gB0 = B + (size_t)(nBase + rS) * ldb + seg;

    auto stage = [&](int k0, int buf) {
        const u32 lA = (u32)(size_t)&As[buf][rS * 40 + seg];
        const u32 lB = (u32)(size_t)&Bs[buf][rS * 40 + seg];
        const u16* gA = gA0 + k0;
        const u16* gB = gB0 + k0;
        asm volatile("global_load_async_to_lds_b128 %0, %1, off"
                     :: "v"(lA), "v"(gA) : "memory");
        asm volatile("global_load_async_to_lds_b128 %0, %1, off offset:16"
                     :: "v"(lA), "v"(gA) : "memory");
        asm volatile("global_load_async_to_lds_b128 %0, %1, off"
                     :: "v"(lB), "v"(gB) : "memory");
        asm volatile("global_load_async_to_lds_b128 %0, %1, off offset:16"
                     :: "v"(lB), "v"(gB) : "memory");
    };

    v8f zero = {};
    v8f acc[2][4];
    #pragma unroll
    for (int mt = 0; mt < 2; ++mt)
        #pragma unroll
        for (int nt = 0; nt < 4; ++nt)
            acc[mt][nt] = zero;

    stage(0, 0);
    const int nk = K >> 5;
    for (int kt = 0; kt < nk; ++kt) {
        const int cur = kt & 1;
        // My async writes to buf[cur] done -> then block-wide barrier.
        asm volatile("s_wait_asynccnt 0x0" ::: "memory");
        __syncthreads();
        if (kt + 1 < nk) stage((kt + 1) << 5, cur ^ 1);

        // Fragments: two ds_load_b128 each.
        // Chunk 1 at byte 80*r + 16*hi, chunk 2 at byte 80*r + 32 + 16*hi.
        const uint4* AsQ = (const uint4*)As[cur];
        const uint4* BsQ = (const uint4*)Bs[cur];

        FragU af[2];
        #pragma unroll
        for (int mt = 0; mt < 2; ++mt) {
            const int r = wm*32 + mt*16 + lo;
            af[mt].q[0] = AsQ[5*r + hi];
            af[mt].q[1] = AsQ[5*r + 2 + hi];
        }
        FragU bfr[4];
        #pragma unroll
        for (int nt = 0; nt < 4; ++nt) {
            const int r = wn*64 + nt*16 + lo;
            bfr[nt].q[0] = BsQ[5*r + hi];
            bfr[nt].q[1] = BsQ[5*r + 2 + hi];
        }

        #pragma unroll
        for (int mt = 0; mt < 2; ++mt)
            #pragma unroll
            for (int nt = 0; nt < 4; ++nt)
                acc[mt][nt] = __builtin_amdgcn_wmma_f32_16x16x32_bf16(
                    false, af[mt].v, false, bfr[nt].v,
                    (short)0, acc[mt][nt], false, false);
    }

    // Epilogue. C/D layout: VGPR r -> row (hi ? 8 : 0)+r, col = lo
    #pragma unroll
    for (int mt = 0; mt < 2; ++mt) {
        #pragma unroll
        for (int nt = 0; nt < 4; ++nt) {
            #pragma unroll
            for (int r = 0; r < 8; ++r) {
                const int row = mBase + wm*32 + mt*16 + hi*8 + r;
                const int col = nBase + wn*64 + nt*16 + lo;
                const size_t idx = cbase + (size_t)row * ldc + col;
                float v = acc[mt][nt][r];
                if constexpr (EPI == 0) {
                    ((u16*)Cv)[idx] = f32_to_bf16(v);
                } else if constexpr (EPI == 1) {
                    ((u16*)Cv)[idx] = f32_to_bf16(gelu_exact(v + bias[col]));
                } else if constexpr (EPI == 2) {
                    ((u16*)Cv)[idx] = f32_to_bf16(v + bias[col]);
                } else {
                    ((float*)Cv)[idx] =
                        v + resid[(size_t)batch * sR + (size_t)row * ldc + col];
                }
            }
        }
    }
}

// ---------------------------------------------------------------------------
extern "C" void kernel_launch(void* const* d_in, const int* in_sizes, int n_in,
                              void* d_out, int out_size, void* d_ws, size_t ws_size,
                              hipStream_t stream) {
    (void)in_sizes; (void)n_in; (void)out_size; (void)ws_size;
    const float* x     = (const float*)d_in[0];
    const float* gamma = (const float*)d_in[1];
    const float* beta  = (const float*)d_in[2];
    const float* w1    = (const float*)d_in[3];
    const float* b1    = (const float*)d_in[4];
    const float* w2    = (const float*)d_in[5];
    const float* b2    = (const float*)d_in[6];

    constexpr long long BSZ = 4, T = 2048, DIM = 1024, H = 2048;
    constexpr long long M = BSZ * T;  // 8192

    char* ws = (char*)d_ws;
    size_t off = 0;
    auto take = [&](size_t bytes) {
        char* p = ws + off;
        off = (off + bytes + 255) & ~(size_t)255;
        return p;
    };
    u16*   xb  = (u16*)take((size_t)M * DIM * 2);     // bf16(x)
    u16*   xn  = (u16*)take((size_t)M * DIM * 2);     // bf16(LN(x))
    u16*   w1t = (u16*)take((size_t)H * DIM * 2);     // w1^T  [H,DIM]
    u16*   w2t = (u16*)take((size_t)DIM * H * 2);     // w2^T  [DIM,H]
    u16*   y1  = (u16*)take((size_t)M * H * 2);       // gelu(xn@w1+b1)
    u16*   y2  = (u16*)take((size_t)M * DIM * 2);     // y1@w2+b2
    u16*   y2t = (u16*)take((size_t)M * DIM * 2);     // y2^T per batch [DIM,T]
    u16*   sc  = (u16*)take((size_t)BSZ * T * T * 2); // x@x^T
    float* yp  = (float*)take((size_t)M * DIM * 4);   // sc@y2 + x

    // 1) LN + bf16 conversions (weights converted AND transposed to K-major)
    ln_in_kernel<<<dim3((unsigned)M), dim3(256), 0, stream>>>(x, gamma, beta, xb, xn);
    convert_transpose_kernel<<<dim3((unsigned)(H/32), (unsigned)(DIM/32)), dim3(32, 8), 0, stream>>>(
        w1, w1t, (int)DIM, (int)H);
    convert_transpose_kernel<<<dim3((unsigned)(DIM/32), (unsigned)(H/32)), dim3(32, 8), 0, stream>>>(
        w2, w2t, (int)H, (int)DIM);

    // 2) y1 = gelu(xn @ w1 + b1)    A[8192,1024] x B^T(w1t[2048,1024])
    gemm_bf16_kernel<1><<<dim3((unsigned)(H/128), (unsigned)(M/128), 1), 256, 0, stream>>>(
        xn, (int)DIM, 0, w1t, (int)DIM, 0, y1, (int)H, 0, b1, nullptr, 0, (int)DIM);

    // 3) y2 = y1 @ w2 + b2          A[8192,2048] x B^T(w2t[1024,2048])
    gemm_bf16_kernel<2><<<dim3((unsigned)(DIM/128), (unsigned)(M/128), 1), 256, 0, stream>>>(
        y1, (int)H, 0, w2t, (int)H, 0, y2, (int)DIM, 0, b2, nullptr, 0, (int)H);

    // 4) scores = x @ x^T per batch  (B = xb, naturally K-major)
    gemm_bf16_kernel<0><<<dim3((unsigned)(T/128), (unsigned)(T/128), (unsigned)BSZ), 256, 0, stream>>>(
        xb, (int)DIM, T*DIM, xb, (int)DIM, T*DIM, sc, (int)T, T*T, nullptr, nullptr, 0, (int)DIM);

    // 4b) y2t[b] = y2[b]^T  ([T,DIM] -> [DIM,T] per batch)
    transpose_u16_kernel<<<dim3((unsigned)(DIM/32), (unsigned)(T/32), (unsigned)BSZ), dim3(32, 8), 0, stream>>>(
        y2, y2t, (int)T, (int)DIM, T*DIM, T*DIM);

    // 5) yp = scores @ y2 + x       A(sc[T,T]) x B^T(y2t[DIM,T]) + residual
    gemm_bf16_kernel<3><<<dim3((unsigned)(DIM/128), (unsigned)(T/128), (unsigned)BSZ), 256, 0, stream>>>(
        sc, (int)T, T*T, y2t, (int)T, T*DIM, yp, (int)DIM, T*DIM, nullptr, x, T*DIM, (int)T);

    // 6) out = LN(yp)
    ln_out_kernel<<<dim3((unsigned)M), 256, 0, stream>>>(yp, gamma, beta, (float*)d_out);
}
